// DeepSeekMultiHeadAttention_28965259444957
// MI455X (gfx1250) — matrix operand order repaired and checked by reading, not compiled, not run on hardware
//
#include <hip/hip_runtime.h>

// ---------------------------------------------------------------------------
// DeepSeek MLA forward on gfx1250 (MI455X), bf16 WMMA everywhere.
// B=1, T=2048, D=2048, H=16, DK=128 (64 nope + 64 rope), V=128, KVR=512, QR=768
//
// GEMMs: register-blocked (32x64 C per wave) + async LDS staging of B slabs
//        via global_load_async_to_lds_b128 / s_wait_asynccnt (CDNA5 ASYNCcnt).
// Attention: flash-style, S^T = K*Q^T trick so the score tile layout equals
//        the A-fragment layout of the P*V WMMA (no cross-lane transpose);
//        K/V slabs async-staged into LDS and shared by all 8 waves of a block.
// ---------------------------------------------------------------------------

#define T_SEQ   2048
#define D_MODEL 2048
#define NHEADS  16
#define DKQ     128
#define NOPE_D  64
#define ROPE_D  64
#define VDIM    128
#define KVR_D   512
#define QR_D    768
#define CKV_D   (KVR_D + ROPE_D)              // 576
#define KVB_D   (NHEADS * (NOPE_D + VDIM))    // 3072
#define QOUT_D  (NHEADS * DKQ)                // 2048
#define SCALE_F 0.08838834764831845f          // 128^-0.5
#define EPS_F   1.1920929e-07f                // np.finfo(float32).eps
#define LN10000 9.210340371976184f            // ln(10000)

#define BN_BLK  256                           // GEMM block N coverage
#define BK      32                            // GEMM K step (WMMA depth)

typedef __bf16 bf16_t;
typedef __attribute__((ext_vector_type(16))) __bf16 v16bf;
typedef __attribute__((ext_vector_type(8)))  __bf16 v8bf;
typedef __attribute__((ext_vector_type(8)))  float  v8f;

// fp32 -> bf16 with round-to-nearest-even; __bf16 used as storage only.
__device__ __forceinline__ bf16_t f2bf(float f) {
    union { float f; unsigned u; } in; in.f = f;
    unsigned u = in.u;
    u += 0x7FFFu + ((u >> 16) & 1u);
    union { unsigned short s; bf16_t b; } out;
    out.s = (unsigned short)(u >> 16);
    return out.b;
}

// A-fragment loader for V_WMMA_F32_16X16X32_BF16 (ISA 7.12.2, 16-bit A 16x32):
// lane%16 = M row; elements j<8 -> K = half*8 + j ; j>=8 -> K = half*8 + 8 + j.
// Works for both global and LDS source pointers (16B-aligned chunks).
__device__ __forceinline__ v16bf load_a_frag(const bf16_t* row_k, int half) {
    v8bf lo = *(const v8bf*)(row_k + half * 8);
    v8bf hi = *(const v8bf*)(row_k + half * 8 + 16);
    v16bf a;
#pragma unroll
    for (int i = 0; i < 8; ++i) { a[i] = lo[i]; a[i + 8] = hi[i]; }
    return a;
}

// CDNA5 async copy: global -> LDS, 16 bytes per lane, tracked by ASYNCcnt.
// VDST = LDS byte offset (low 32 bits of the flat shared address), VADDR = global.
__device__ __forceinline__ void async_g2l_b128(unsigned lds_off, const void* gaddr) {
    asm volatile("global_load_async_to_lds_b128 %0, %1, off"
                 :: "v"(lds_off), "v"(gaddr) : "memory");
}
__device__ __forceinline__ void wait_asynccnt0() {
    asm volatile("s_wait_asynccnt 0x0" ::: "memory");
}

// ---------------------------------------------------------------------------
// Elementwise fp32 -> bf16 conversion
// ---------------------------------------------------------------------------
__global__ void cvt_f32_bf16(const float* __restrict__ s, bf16_t* __restrict__ d, int n) {
    int i = blockIdx.x * blockDim.x + threadIdx.x;
    if (i < n) d[i] = f2bf(s[i]);
}

// ---------------------------------------------------------------------------
// Blocked bf16 GEMM: C[M,N](f32) = A[M,K](bf16 rm) x B[K,N](bf16 rm)
// Requires M%64==0 and N%64==0 (true for every call site here).
// Block = 256 threads = 8 waves arranged 2(M)x4(N): macro tile 64M x 256N.
// Each wave owns a 32x64 C tile = 2x4 WMMA accumulators (8 wmma per K step).
// ---------------------------------------------------------------------------
__device__ __forceinline__ void stage_b_slab(const bf16_t* __restrict__ B, int N,
                                             int nblk, int k, bf16_t* slab, int tid) {
    // 32 rows x 512 B/row = 1024 x 16B chunks; 4 chunks per thread.
#pragma unroll
    for (int c = 0; c < 4; ++c) {
        int ci    = tid * 4 + c;        // 0..1023
        int row   = ci >> 5;            // 0..31   (K offset)
        int col16 = ci & 31;            // 16B chunk within row
        int gcol  = nblk + col16 * 8;   // element column in B
        if (gcol < N) {
            unsigned lds_off = (unsigned)(size_t)(slab + row * BN_BLK + col16 * 8);
            const void* g = (const void*)(B + (size_t)(k + row) * N + gcol);
            async_g2l_b128(lds_off, g);
        }
    }
}

__global__ void gemm_bf16_f32(const bf16_t* __restrict__ A, const bf16_t* __restrict__ B,
                              float* __restrict__ C, int M, int N, int K) {
    __shared__ __attribute__((aligned(16))) bf16_t bs[2][BK * BN_BLK]; // 2 x 16 KB

    const int tid  = threadIdx.x;
    const int wave = tid >> 5;
    const int lane = tid & 31;
    const int r    = lane & 15;
    const int half = lane >> 4;
    const int wm   = wave >> 2;              // 0..1 -> M sub-tile
    const int wn   = wave & 3;               // 0..3 -> N sub-tile
    const int m0   = blockIdx.y * 64 + wm * 32;
    const int nblk = blockIdx.x * BN_BLK;
    const int n0   = nblk + wn * 64;
    const bool active = (n0 < N);            // wave-uniform (N%64==0 tail guard)

    const bf16_t* arow0 = A + (size_t)(m0 + r) * K;
    const bf16_t* arow1 = A + (size_t)(m0 + 16 + r) * K;

    const v8f zero8 = {0.f, 0.f, 0.f, 0.f, 0.f, 0.f, 0.f, 0.f};
    v8f acc[2][4];
#pragma unroll
    for (int mi = 0; mi < 2; ++mi)
#pragma unroll
        for (int nt = 0; nt < 4; ++nt) acc[mi][nt] = zero8;

    stage_b_slab(B, N, nblk, 0, &bs[0][0], tid);

    int buf = 0;
    for (int k = 0; k < K; k += BK) {
        wait_asynccnt0();                    // my async writes to bs[buf] landed
        __syncthreads();                     // everyone's writes visible
        if (k + BK < K)
            stage_b_slab(B, N, nblk, k + BK, &bs[buf ^ 1][0], tid);

        if (active) {
            if (k + 4 * BK < K) __builtin_prefetch(arow0 + k + 4 * BK, 0, 1);
            v16bf a0 = load_a_frag(arow0 + k, half);
            v16bf a1 = load_a_frag(arow1 + k, half);
            const bf16_t* brow = &bs[buf][wn * 64 + r]; // lane%16 = N column, elements = K (table order)
            v16bf bfr[4];
#pragma unroll
            for (int nt = 0; nt < 4; ++nt)
#pragma unroll
                for (int j = 0; j < 16; ++j)
                    bfr[nt][j] = brow[((j & 7) + 8 * half + ((j >> 3) << 4)) * BN_BLK + nt * 16];
#pragma unroll
            for (int nt = 0; nt < 4; ++nt) {
                acc[0][nt] = __builtin_amdgcn_wmma_f32_16x16x32_bf16(
                    false, a0, false, bfr[nt], (short)0, acc[0][nt], false, false);
                acc[1][nt] = __builtin_amdgcn_wmma_f32_16x16x32_bf16(
                    false, a1, false, bfr[nt], (short)0, acc[1][nt], false, false);
            }
        }
        __syncthreads();                     // reads done before slab reuse
        buf ^= 1;
    }

    if (active) {
#pragma unroll
        for (int mi = 0; mi < 2; ++mi)
#pragma unroll
            for (int nt = 0; nt < 4; ++nt)
#pragma unroll
                for (int v = 0; v < 8; ++v)
                    C[(size_t)(m0 + mi * 16 + v + 8 * half) * N + n0 + nt * 16 + r] =
                        acc[mi][nt][v];
    }
}

// ---------------------------------------------------------------------------
// RMS-norm one row per block (fp32 in, bf16 out)
// ---------------------------------------------------------------------------
__global__ void rmsnorm_f32_bf16(const float* __restrict__ src, const float* __restrict__ w,
                                 bf16_t* __restrict__ dst, int stride, int cols) {
    __shared__ float red[256];
    const int row = blockIdx.x;
    const int tid = threadIdx.x;
    const float* s = src + (size_t)row * stride;
    float ss = 0.f;
    for (int c = tid; c < cols; c += 256) { float v = s[c]; ss += v * v; }
    red[tid] = ss;
    __syncthreads();
    for (int o = 128; o > 0; o >>= 1) {
        if (tid < o) red[tid] += red[tid + o];
        __syncthreads();
    }
    const float inv = rsqrtf(red[0] / (float)cols + EPS_F);
    for (int c = tid; c < cols; c += 256)
        dst[(size_t)row * cols + c] = f2bf(s[c] * inv * w[c]);
}

// ---------------------------------------------------------------------------
// RoPE helpers + head-major Q/K/V assembly (fp32 math, bf16 out)
// ---------------------------------------------------------------------------
__device__ __forceinline__ void rope_cs(int t, int i, float* c, float* s) {
    float freq = __expf(-((float)(2 * i) * (1.0f / (float)ROPE_D)) * LN10000);
    float ang  = (float)t * freq;
    __sincosf(ang, s, c);
}

// q_f [T, H*128] -> qs [H][T][128], rope applied to last 64 dims
__global__ void build_qs(const float* __restrict__ qf, bf16_t* __restrict__ qs) {
    const int t = blockIdx.x;
    for (int hd = threadIdx.x; hd < NHEADS * DKQ; hd += blockDim.x) {
        int h = hd >> 7, d = hd & 127;
        const float* qrow = qf + (size_t)t * QOUT_D + h * DKQ;
        float val;
        if (d < NOPE_D) {
            val = qrow[d];
        } else {
            int j = d - NOPE_D;
            int i = (j < 32) ? j : (j - 32);
            float c, s; rope_cs(t, i, &c, &s);
            float x1 = qrow[NOPE_D + i];
            float x2 = qrow[NOPE_D + 32 + i];
            val = (j < 32) ? (x1 * c - x2 * s) : (x1 * s + x2 * c);
        }
        qs[((size_t)h * T_SEQ + t) * DKQ + d] = f2bf(val);
    }
}

// kv_f [T, H*192] + ckv_f [T,576] (k_pe at col 512) -> ks [H][T][128], v [H][T][128]
__global__ void build_ksv(const float* __restrict__ kvf, const float* __restrict__ ckv,
                          bf16_t* __restrict__ ks, bf16_t* __restrict__ vv) {
    const int t = blockIdx.x;
    const float* kvrow = kvf + (size_t)t * KVB_D;
    const float* pe    = ckv + (size_t)t * CKV_D + KVR_D;
    for (int hd = threadIdx.x; hd < NHEADS * DKQ; hd += blockDim.x) {
        int h = hd >> 7, d = hd & 127;
        float kval;
        if (d < NOPE_D) {
            kval = kvrow[h * (NOPE_D + VDIM) + d];
        } else {
            int j = d - NOPE_D;
            int i = (j < 32) ? j : (j - 32);
            float c, s; rope_cs(t, i, &c, &s);
            float x1 = pe[i], x2 = pe[32 + i];
            kval = (j < 32) ? (x1 * c - x2 * s) : (x1 * s + x2 * c);
        }
        ks[((size_t)h * T_SEQ + t) * DKQ + d] = f2bf(kval);
        vv[((size_t)h * T_SEQ + t) * VDIM + d] = f2bf(kvrow[h * (NOPE_D + VDIM) + NOPE_D + d]);
    }
}

// ---------------------------------------------------------------------------
// Flash attention, one wave per 16-query tile per head; 8 waves per block all
// scan the same head's K/V stream, so 32-key K/V slabs (8 KB + 8 KB) are
// async-staged into LDS once per block and consumed by all waves.
// S^T = K(16x128) x Q^T(128x16): C layout (lane%16=query, element v -> key
// v+8*half) is exactly the A-fragment layout for P x V over a 32-key chunk.
// ---------------------------------------------------------------------------
__device__ __forceinline__ void stage_kv_slab(const bf16_t* __restrict__ kh,
                                              const bf16_t* __restrict__ vh,
                                              int kb, bf16_t* kslab, bf16_t* vslab,
                                              int tid) {
    // Each slab: 32 rows x 256 B = 512 x 16B chunks; 2 K-chunks + 2 V-chunks per thread.
#pragma unroll
    for (int c = 0; c < 2; ++c) {
        int ci    = tid * 2 + c;        // 0..511
        int row   = ci >> 4;            // 0..31 (key offset)
        int col16 = ci & 15;            // 16B chunk within row
        int eoff  = row * DKQ + col16 * 8;
        async_g2l_b128((unsigned)(size_t)(kslab + eoff),
                       (const void*)(kh + (size_t)(kb + row) * DKQ + col16 * 8));
        async_g2l_b128((unsigned)(size_t)(vslab + eoff),
                       (const void*)(vh + (size_t)(kb + row) * VDIM + col16 * 8));
    }
}

__global__ void attn_kernel(const bf16_t* __restrict__ qs, const bf16_t* __restrict__ ks,
                            const bf16_t* __restrict__ vv, bf16_t* __restrict__ ob) {
    __shared__ __attribute__((aligned(16))) bf16_t kslab[2][32 * DKQ];  // 2 x 8 KB
    __shared__ __attribute__((aligned(16))) bf16_t vslab[2][32 * VDIM]; // 2 x 8 KB

    const int tid  = threadIdx.x;
    const int wave = tid >> 5;
    const int lane = tid & 31;
    const int r    = lane & 15;
    const int half = lane >> 4;
    const int h    = blockIdx.y;
    const int q0   = (blockIdx.x * 8 + wave) * 16;

    const bf16_t* qh = qs + (size_t)h * T_SEQ * DKQ;
    const bf16_t* kh = ks + (size_t)h * T_SEQ * DKQ;
    const bf16_t* vh = vv + (size_t)h * T_SEQ * VDIM;

    // Q^T B-fragments (lane%16 = query, element j = contraction offset in table order), 4 K-chunks.
    v16bf bq[4];
#pragma unroll
    for (int kc = 0; kc < 4; ++kc) {
#pragma unroll
        for (int j = 0; j < 16; ++j)
            bq[kc][j] = qh[(size_t)(q0 + r) * DKQ + kc * 32 + (j & 7) + 8 * half + ((j >> 3) << 4)];
    }

    const v8f zero8 = {0.f, 0.f, 0.f, 0.f, 0.f, 0.f, 0.f, 0.f};
    v8f acc[8];
#pragma unroll
    for (int dt = 0; dt < 8; ++dt) acc[dt] = zero8;

    float m_cur = -1e30f;
    float l     = 0.f;

    stage_kv_slab(kh, vh, 0, &kslab[0][0], &vslab[0][0], tid);

    int buf = 0;
    for (int kb = 0; kb < T_SEQ; kb += 32) {
        wait_asynccnt0();
        __syncthreads();
        if (kb + 32 < T_SEQ)
            stage_kv_slab(kh, vh, kb + 32, &kslab[buf ^ 1][0], &vslab[buf ^ 1][0], tid);

        // S^T tiles for keys [kb, kb+16) and [kb+16, kb+32), K fragments from LDS.
        v8f s0 = zero8, s1 = zero8;
        const bf16_t* k0row = &kslab[buf][r * DKQ];
        const bf16_t* k1row = &kslab[buf][(16 + r) * DKQ];
#pragma unroll
        for (int kc = 0; kc < 4; ++kc) {
            v16bf a0 = load_a_frag(k0row + kc * 32, half);
            s0 = __builtin_amdgcn_wmma_f32_16x16x32_bf16(false, a0, false, bq[kc],
                                                         (short)0, s0, false, false);
        }
#pragma unroll
        for (int kc = 0; kc < 4; ++kc) {
            v16bf a1 = load_a_frag(k1row + kc * 32, half);
            s1 = __builtin_amdgcn_wmma_f32_16x16x32_bf16(false, a1, false, bq[kc],
                                                         (short)0, s1, false, false);
        }

        // Online softmax (fp32). Lane holds query q=r; keys split across halves.
        float sv0[8], sv1[8];
        float mt = -1e30f;
#pragma unroll
        for (int v = 0; v < 8; ++v) {
            sv0[v] = s0[v] * SCALE_F;
            sv1[v] = s1[v] * SCALE_F;
            mt = fmaxf(mt, fmaxf(sv0[v], sv1[v]));
        }
        mt = fmaxf(mt, __shfl_xor(mt, 16, 32));
        const float m_new = fmaxf(m_cur, mt);

        float p0[8], p1[8], ls = 0.f;
#pragma unroll
        for (int v = 0; v < 8; ++v) {
            p0[v] = __expf(sv0[v] - m_new);
            p1[v] = __expf(sv1[v] - m_new);
            ls += p0[v] + p1[v];
        }
        ls += __shfl_xor(ls, 16, 32);
        const float alpha = __expf(m_cur - m_new);
        m_cur = m_new;
        l = l * alpha + ls;

        float fq[8];
#pragma unroll
        for (int v = 0; v < 8; ++v) fq[v] = __shfl(alpha, v + 8 * half, 32);
#pragma unroll
        for (int dt = 0; dt < 8; ++dt) {
#pragma unroll
            for (int v = 0; v < 8; ++v) acc[dt][v] *= fq[v];
        }

        // P A-fragment: elements 0..7 from S^T tile 0, 8..15 from tile 1.
        v16bf pa;
#pragma unroll
        for (int j = 0; j < 8; ++j) { pa[j] = f2bf(p0[j]); pa[j + 8] = f2bf(p1[j]); }

        // P x V: B-fragment lane%16 = value dim, element j = key offset (table order); from LDS.
        const bf16_t* vrow = &vslab[buf][r];
#pragma unroll
        for (int dt = 0; dt < 8; ++dt) {
            v16bf bv;
#pragma unroll
            for (int j = 0; j < 16; ++j)
                bv[j] = vrow[((j & 7) + 8 * half + ((j >> 3) << 4)) * VDIM + dt * 16];
            acc[dt] = __builtin_amdgcn_wmma_f32_16x16x32_bf16(false, pa, false, bv,
                                                              (short)0, acc[dt], false, false);
        }

        __syncthreads();                 // reads done before slab reuse
        buf ^= 1;
    }

    float linv[8];
#pragma unroll
    for (int v = 0; v < 8; ++v) {
        float lq = __shfl(l, v + 8 * half, 32);
        linv[v] = 1.0f / lq;
    }
#pragma unroll
    for (int dt = 0; dt < 8; ++dt) {
#pragma unroll
        for (int v = 0; v < 8; ++v)
            ob[(size_t)(q0 + v + 8 * half) * QOUT_D + h * VDIM + dt * 16 + r] =
                f2bf(acc[dt][v] * linv[v]);
    }
}

// ---------------------------------------------------------------------------
// Host-side orchestration
// ---------------------------------------------------------------------------
extern "C" void kernel_launch(void* const* d_in, const int* in_sizes, int n_in,
                              void* d_out, int out_size, void* d_ws, size_t ws_size,
                              hipStream_t stream) {
    (void)in_sizes; (void)n_in; (void)out_size; (void)ws_size;

    const float* x           = (const float*)d_in[0]; // [T, D]
    const float* wq_a        = (const float*)d_in[1]; // [D, QR]
    const float* q_a_norm_w  = (const float*)d_in[2]; // [QR]
    const float* wq_b        = (const float*)d_in[3]; // [QR, H*DK]
    const float* wkv_a       = (const float*)d_in[4]; // [D, KVR+ROPE]
    const float* kv_a_norm_w = (const float*)d_in[5]; // [KVR]
    const float* wkv_b       = (const float*)d_in[6]; // [KVR, H*192]
    const float* wo          = (const float*)d_in[7]; // [H*V, D]
    float*       out         = (float*)d_out;         // [T, D]

    char* ws = (char*)d_ws;
    size_t off = 0;
    auto take = [&](size_t bytes) -> char* {
        char* p = ws + off;
        off += (bytes + 255) & ~(size_t)255;
        return p;
    };

    bf16_t* xb    = (bf16_t*)take((size_t)T_SEQ * D_MODEL * 2);
    bf16_t* wqab  = (bf16_t*)take((size_t)D_MODEL * QR_D * 2);
    bf16_t* wqbb  = (bf16_t*)take((size_t)QR_D * QOUT_D * 2);
    bf16_t* wkvab = (bf16_t*)take((size_t)D_MODEL * CKV_D * 2);
    bf16_t* wkvbb = (bf16_t*)take((size_t)KVR_D * KVB_D * 2);
    bf16_t* wob   = (bf16_t*)take((size_t)QOUT_D * D_MODEL * 2);
    float*  qa    = (float*) take((size_t)T_SEQ * QR_D * 4);
    float*  ckv   = (float*) take((size_t)T_SEQ * CKV_D * 4);
    bf16_t* qn    = (bf16_t*)take((size_t)T_SEQ * QR_D * 2);
    bf16_t* cn    = (bf16_t*)take((size_t)T_SEQ * KVR_D * 2);
    float*  qf    = (float*) take((size_t)T_SEQ * QOUT_D * 4);
    float*  kvf   = (float*) take((size_t)T_SEQ * KVB_D * 4);
    bf16_t* qsb   = (bf16_t*)take((size_t)NHEADS * T_SEQ * DKQ * 2);
    bf16_t* ksb   = (bf16_t*)take((size_t)NHEADS * T_SEQ * DKQ * 2);
    bf16_t* vb    = (bf16_t*)take((size_t)NHEADS * T_SEQ * VDIM * 2);
    bf16_t* obb   = (bf16_t*)take((size_t)T_SEQ * QOUT_D * 2);

    auto cvt = [&](const float* s, bf16_t* d, int n) {
        cvt_f32_bf16<<<(n + 255) / 256, 256, 0, stream>>>(s, d, n);
    };

    // 1) bf16 conversions (x + all weights)
    cvt(x,     xb,    T_SEQ * D_MODEL);
    cvt(wq_a,  wqab,  D_MODEL * QR_D);
    cvt(wq_b,  wqbb,  QR_D * QOUT_D);
    cvt(wkv_a, wkvab, D_MODEL * CKV_D);
    cvt(wkv_b, wkvbb, KVR_D * KVB_D);
    cvt(wo,    wob,   QOUT_D * D_MODEL);

    auto gemm = [&](const bf16_t* A, const bf16_t* B, float* C, int M, int N, int K) {
        gemm_bf16_f32<<<dim3((N + BN_BLK - 1) / BN_BLK, M / 64), 256, 0, stream>>>(
            A, B, C, M, N, K);
    };

    // 2) q_a = x @ wq_a ; ckv = x @ wkv_a
    gemm(xb, wqab,  qa,  T_SEQ, QR_D,  D_MODEL);
    gemm(xb, wkvab, ckv, T_SEQ, CKV_D, D_MODEL);

    // 3) RMS norms -> bf16 activations
    rmsnorm_f32_bf16<<<T_SEQ, 256, 0, stream>>>(qa,  q_a_norm_w,  qn, QR_D,  QR_D);
    rmsnorm_f32_bf16<<<T_SEQ, 256, 0, stream>>>(ckv, kv_a_norm_w, cn, CKV_D, KVR_D);

    // 4) q = qn @ wq_b ; kv = cn @ wkv_b
    gemm(qn, wqbb,  qf,  T_SEQ, QOUT_D, QR_D);
    gemm(cn, wkvbb, kvf, T_SEQ, KVB_D,  KVR_D);

    // 5) RoPE + head-major Q/K/V assembly
    build_qs <<<T_SEQ, 256, 0, stream>>>(qf, qsb);
    build_ksv<<<T_SEQ, 256, 0, stream>>>(kvf, ckv, ksb, vb);

    // 6) Flash attention: grid (T/128 query-tiles-of-8-waves, H)
    attn_kernel<<<dim3(T_SEQ / 128, NHEADS), 256, 0, stream>>>(qsb, ksb, vb, obb);

    // 7) out = attn_out @ wo (fp32 result)
    gemm(obb, wob, out, T_SEQ, D_MODEL, QOUT_D);
}
